// MultiheadAttention_69578470195934
// MI455X (gfx1250) — matrix operand order repaired and checked by reading, not compiled, not run on hardware
//
#include <hip/hip_runtime.h>
#include <hip/hip_bf16.h>
#include <hip/hip_fp16.h>

// ---------------------------------------------------------------------------
// MHA for MI455X (gfx1250): QKV proj (relu6) in f16 WMMA + flash attention.
// v2: wider tiles for WMMA issue density (proj 128x64 / K64, attn key tile 64).
// ---------------------------------------------------------------------------

typedef __attribute__((ext_vector_type(16))) _Float16 v16h;
typedef __attribute__((ext_vector_type(8)))  _Float16 v8h;
typedef __attribute__((ext_vector_type(4)))  _Float16 v4h;
typedef __attribute__((ext_vector_type(8)))  float    v8f;

#define HIDDEN    1024
#define NUM_HEADS 16
#define HEAD      64
#define TT        2048
#define BATCH     4
#define MTOT      (TT * BATCH)          // 8192 tokens
#define TOKSTRIDE (BATCH * HIDDEN)      // 4096 elems between consecutive t
#define S_TILE    64                    // keys per flash iteration

// ---------------------------------------------------------------------------
// WMMA helper (CDNA5 16x16x32 f16 -> f32 accumulate)
// ---------------------------------------------------------------------------
__device__ __forceinline__ v8f wmma_f16(v16h a, v16h b, v8f c) {
    return __builtin_amdgcn_wmma_f32_16x16x32_f16(
        /*neg_a=*/false, a, /*neg_b=*/false, b,
        /*c_mod=*/(short)0, c, /*reuse_a=*/false, /*reuse_b=*/false);
}

// A-matrix fragment (16x32 f16, row-major source, leading dim in elements).
// ISA 7.12.2: lane<16 holds row m=lane, K={0..7,16..23}; lane>=16 holds
// m=lane-16, K={8..15,24..31}.
__device__ __forceinline__ v16h load_a_frag(const _Float16* base, int ldm, int lane) {
    const int m  = lane & 15;
    const int ks = (lane >> 4) << 3;          // 0 or 8
    const _Float16* p = base + (size_t)m * ldm + ks;
    union { v16h v; v8h h[2]; } u;
    u.h[0] = *reinterpret_cast<const v8h*>(p);        // K = ks   .. ks+7
    u.h[1] = *reinterpret_cast<const v8h*>(p + 16);   // K = ks+16.. ks+23
    return u.v;
}

// B-matrix fragment (32x16 f16) where B[k][n] = Src[n][k]:
// source rows are the N dimension, contiguous along K.
// lane<16: n=lane, K={0..7,16..23} ; lane>=16: n=lane-16, K={8..15,24..31}.
__device__ __forceinline__ v16h load_b_frag_nk(const _Float16* base, int ldn, int lane) {
    const int n  = lane & 15;
    const int kb = (lane >> 4) << 3;          // 0 or 8
    const _Float16* p = base + (size_t)n * ldn + kb;
    union { v16h v; v8h h[2]; } u;
    u.h[0] = *reinterpret_cast<const v8h*>(p);
    u.h[1] = *reinterpret_cast<const v8h*>(p + 16);
    return u.v;
}

// ---------------------------------------------------------------------------
// Kernel 1: Y = (f16) relu6(X @ W^T + b)
// Block: 256 thr = 8 waves; block tile M=128, N=64; K chunk 64 via LDS.
// Wave w owns rows [w*16, w*16+16); computes 4 N-tiles -> 8 WMMAs / chunk.
// ---------------------------------------------------------------------------
__global__ __launch_bounds__(256)
void proj_relu6_kernel(const float* __restrict__ X, const float* __restrict__ W,
                       const float* __restrict__ bias, _Float16* __restrict__ Y) {
    __shared__ __align__(16) _Float16 As[128 * 64];   // [m][k]  16 KB
    __shared__ __align__(16) _Float16 Bs[64 * 64];    // [n][k]   8 KB

    const int tid  = threadIdx.x;
    const int lane = tid & 31;
    const int wave = tid >> 5;                // 0..7
    const int m0   = blockIdx.x * 128;
    const int n0   = blockIdx.y * 64;

    v8f acc[4] = {{}, {}, {}, {}};

    for (int kk = 0; kk < HIDDEN; kk += 64) {
        __syncthreads();                      // previous chunk's frags consumed
        // ---- stage A tile: 128x64 f32 -> f16 (2048 float4 / block, 8/thr) ----
        #pragma unroll
        for (int i = 0; i < 8; ++i) {
            const int f  = tid + i * 256;          // float4 id, [0,2048)
            const int r  = f >> 4;                 // row 0..127
            const int c4 = f & 15;                 // float4 col 0..15
            const float4 x = *reinterpret_cast<const float4*>(
                X + (size_t)(m0 + r) * HIDDEN + kk + c4 * 4);
            v4h hv;
            hv[0] = (_Float16)x.x; hv[1] = (_Float16)x.y;
            hv[2] = (_Float16)x.z; hv[3] = (_Float16)x.w;
            *reinterpret_cast<v4h*>(&As[r * 64 + c4 * 4]) = hv;
        }
        // ---- stage B tile: 64x64 of W (rows n, contiguous k), 4/thr ----
        #pragma unroll
        for (int i = 0; i < 4; ++i) {
            const int f  = tid + i * 256;          // float4 id, [0,1024)
            const int r  = f >> 4;                 // n 0..63
            const int c4 = f & 15;
            const float4 x = *reinterpret_cast<const float4*>(
                W + (size_t)(n0 + r) * HIDDEN + kk + c4 * 4);
            v4h hv;
            hv[0] = (_Float16)x.x; hv[1] = (_Float16)x.y;
            hv[2] = (_Float16)x.z; hv[3] = (_Float16)x.w;
            *reinterpret_cast<v4h*>(&Bs[r * 64 + c4 * 4]) = hv;
        }
        __syncthreads();

        const v16h a0 = load_a_frag(&As[wave * 16 * 64], 64, lane);       // k 0..31
        const v16h a1 = load_a_frag(&As[wave * 16 * 64 + 32], 64, lane);  // k 32..63
        #pragma unroll
        for (int j = 0; j < 4; ++j) {
            const v16h b0 = load_b_frag_nk(&Bs[j * 16 * 64], 64, lane);
            const v16h b1 = load_b_frag_nk(&Bs[j * 16 * 64 + 32], 64, lane);
            acc[j] = wmma_f16(a0, b0, acc[j]);
            acc[j] = wmma_f16(a1, b1, acc[j]);
        }
    }

    // ---- epilogue: bias + relu6, store f16 ----
    const int mbase = m0 + wave * 16 + ((lane >> 4) << 3);  // M split per C layout
    #pragma unroll
    for (int j = 0; j < 4; ++j) {
        const int n    = n0 + j * 16 + (lane & 15);
        const float bn = bias[n];
        #pragma unroll
        for (int r = 0; r < 8; ++r) {
            float y = acc[j][r] + bn;
            y = fminf(fmaxf(y, 0.0f), 6.0f);
            Y[(size_t)(mbase + r) * HIDDEN + n] = (_Float16)y;
        }
    }
}

// ---------------------------------------------------------------------------
// Kernel 2: flash attention over f16 Q'/K'/V' (layout (t*4+b)*1024 + h*64 + d).
// One block = one (b,h) x 128 query rows (8 waves x 16 rows).
// Key/value tile of 64 -> 16 WMMAs per online-softmax round.
// ---------------------------------------------------------------------------
__global__ __launch_bounds__(256)
void flash_attn_kernel(const _Float16* __restrict__ Qh, const _Float16* __restrict__ Kh,
                       const _Float16* __restrict__ Vh, float* __restrict__ Out) {
    __shared__ __align__(16) _Float16 Ks[S_TILE * 64];     // K tile [s][d]   8 KB
    __shared__ __align__(16) _Float16 Vt[64 * S_TILE];     // V^T   [d][s]    8 KB
    __shared__ __align__(16) _Float16 Ps[8][16 * S_TILE];  // per-wave P     16 KB

    const int tid  = threadIdx.x;
    const int lane = tid & 31;
    const int wave = tid >> 5;
    const int bh   = blockIdx.y;            // 0..63
    const int b    = bh >> 4;
    const int h    = bh & 15;
    const int q0   = blockIdx.x * 128 + wave * 16;

    const size_t head_off = (size_t)b * HIDDEN + (size_t)h * HEAD;

    // Q fragments (d=0..31, 32..63), pre-scaled by 1/sqrt(HEAD)=0.125 (exact
    // in f16; |q'| <= 6 after relu6, so no denorm risk).
    const _Float16* Qrow = Qh + head_off + (size_t)q0 * TOKSTRIDE;
    v16h aq0 = load_a_frag(Qrow,      TOKSTRIDE, lane);
    v16h aq1 = load_a_frag(Qrow + 32, TOKSTRIDE, lane);
    aq0 = aq0 * (_Float16)0.125f;
    aq1 = aq1 * (_Float16)0.125f;

    // Online softmax state: each lane owns rows m = (lane>=16 ? 8:0)+r.
    float m_i[8], l_i[8];
    #pragma unroll
    for (int r = 0; r < 8; ++r) { m_i[r] = -1e30f; l_i[r] = 0.0f; }
    v8f o[4] = {{}, {}, {}, {}};

    for (int s0 = 0; s0 < TT; s0 += S_TILE) {
        __syncthreads();   // LDS tiles from previous iteration fully consumed

        // ---- stage K (row-major) + V (transposed): 64x64 f16 each ----
        #pragma unroll
        for (int i = 0; i < 2; ++i) {
            const int e  = tid + i * 256;     // v8h unit id, [0,512)
            const int s  = e >> 3;            // 0..63
            const int d0 = (e & 7) * 8;       // 0..56
            const size_t g = (size_t)(s0 + s) * TOKSTRIDE + head_off + d0;
            *reinterpret_cast<v8h*>(&Ks[s * 64 + d0]) =
                *reinterpret_cast<const v8h*>(Kh + g);
            const v8h vv = *reinterpret_cast<const v8h*>(Vh + g);
            #pragma unroll
            for (int j = 0; j < 8; ++j) Vt[(d0 + j) * S_TILE + s] = vv[j];
        }
        if (s0 + S_TILE < TT) {   // prefetch next tile (global_prefetch_b8)
            const int s = tid >> 2;                               // 0..63
            const size_t g = (size_t)(s0 + S_TILE + s) * TOKSTRIDE + head_off
                             + (size_t)(tid & 3) * 16;
            __builtin_prefetch(Kh + g, 0, 3);
            __builtin_prefetch(Vh + g, 0, 3);
        }
        __syncthreads();

        // ---- scores: 16x64 tile = 8 WMMAs (B[k=d][n=s] = Ks[s][d]) ----
        v8f sc[4];
        #pragma unroll
        for (int j = 0; j < 4; ++j) {
            const v16h b0 = load_b_frag_nk(&Ks[j * 16 * 64], 64, lane);       // d 0..31
            const v16h b1 = load_b_frag_nk(&Ks[j * 16 * 64 + 32], 64, lane);  // d 32..63
            v8f c = {};
            c = wmma_f16(aq0, b0, c);
            c = wmma_f16(aq1, b1, c);
            sc[j] = c;
        }

        // ---- online softmax (row reductions across the 16-lane half) ----
        float alpha[8];
        #pragma unroll
        for (int r = 0; r < 8; ++r) {
            float v = fmaxf(fmaxf(sc[0][r], sc[1][r]), fmaxf(sc[2][r], sc[3][r]));
            v = fmaxf(v, __shfl_xor(v, 1, 32));
            v = fmaxf(v, __shfl_xor(v, 2, 32));
            v = fmaxf(v, __shfl_xor(v, 4, 32));
            v = fmaxf(v, __shfl_xor(v, 8, 32));
            const float mn = fmaxf(m_i[r], v);
            alpha[r] = __expf(m_i[r] - mn);
            m_i[r]   = mn;
            float ps = 0.0f;
            #pragma unroll
            for (int j = 0; j < 4; ++j) {
                sc[j][r] = __expf(sc[j][r] - mn);
                ps += sc[j][r];
            }
            ps += __shfl_xor(ps, 1, 32);
            ps += __shfl_xor(ps, 2, 32);
            ps += __shfl_xor(ps, 4, 32);
            ps += __shfl_xor(ps, 8, 32);
            l_i[r] = l_i[r] * alpha[r] + ps;
        }
        #pragma unroll
        for (int r = 0; r < 8; ++r) {
            o[0][r] *= alpha[r]; o[1][r] *= alpha[r];
            o[2][r] *= alpha[r]; o[3][r] *= alpha[r];
        }

        // ---- P: C-layout -> A-layout via per-wave LDS round trip ----
        {
            _Float16* P = &Ps[wave][0];
            const int c  = lane & 15;
            const int mh = (lane >> 4) << 3;
            #pragma unroll
            for (int r = 0; r < 8; ++r) {
                #pragma unroll
                for (int j = 0; j < 4; ++j)
                    P[(mh + r) * S_TILE + j * 16 + c] = (_Float16)sc[j][r];
            }
        }
        // same-wave LDS ops are in-order; compiler inserts s_wait_dscnt
        const v16h ap0 = load_a_frag(&Ps[wave][0],      S_TILE, lane);  // s 0..31
        const v16h ap1 = load_a_frag(&Ps[wave][0] + 32, S_TILE, lane);  // s 32..63

        // ---- O += P @ V : B[k=s][n=d] = Vt[d][s] -> contiguous nk reads ----
        #pragma unroll
        for (int j = 0; j < 4; ++j) {
            const v16h bv0 = load_b_frag_nk(&Vt[j * 16 * S_TILE], S_TILE, lane);
            const v16h bv1 = load_b_frag_nk(&Vt[j * 16 * S_TILE + 32], S_TILE, lane);
            o[j] = wmma_f16(ap0, bv0, o[j]);
            o[j] = wmma_f16(ap1, bv1, o[j]);
        }
    }

    // ---- normalize and store fp32 output (t, b, h*64+d) ----
    const int mh = (lane >> 4) << 3;
    const int dc = lane & 15;
    #pragma unroll
    for (int r = 0; r < 8; ++r) {
        const int t = q0 + mh + r;
        const float inv = 1.0f / l_i[r];
        float* op = Out + (size_t)t * TOKSTRIDE + head_off;
        op[dc]      = o[0][r] * inv;
        op[16 + dc] = o[1][r] * inv;
        op[32 + dc] = o[2][r] * inv;
        op[48 + dc] = o[3][r] * inv;
    }
}

// ---------------------------------------------------------------------------
// Launch
// ---------------------------------------------------------------------------
extern "C" void kernel_launch(void* const* d_in, const int* in_sizes, int n_in,
                              void* d_out, int out_size, void* d_ws, size_t ws_size,
                              hipStream_t stream) {
    (void)in_sizes; (void)n_in; (void)out_size; (void)ws_size;
    const float* q  = (const float*)d_in[0];
    const float* k  = (const float*)d_in[1];
    const float* v  = (const float*)d_in[2];
    const float* Wq = (const float*)d_in[3];
    const float* bq = (const float*)d_in[4];
    const float* Wk = (const float*)d_in[5];
    const float* bk = (const float*)d_in[6];
    const float* Wv = (const float*)d_in[7];
    const float* bv = (const float*)d_in[8];
    float* out = (float*)d_out;

    _Float16* qh = (_Float16*)d_ws;                          // 16 MB each
    _Float16* kh = qh + (size_t)MTOT * HIDDEN;
    _Float16* vh = kh + (size_t)MTOT * HIDDEN;

    dim3 gproj(MTOT / 128, HIDDEN / 64);                     // (64, 16)
    proj_relu6_kernel<<<gproj, 256, 0, stream>>>(q, Wq, bq, qh);
    proj_relu6_kernel<<<gproj, 256, 0, stream>>>(k, Wk, bk, kh);
    proj_relu6_kernel<<<gproj, 256, 0, stream>>>(v, Wv, bv, vh);

    dim3 gattn(TT / 128, BATCH * NUM_HEADS);                 // (16, 64)
    flash_attn_kernel<<<gattn, 256, 0, stream>>>(qh, kh, vh, out);
}
